// ResBlockD2D_47863115546767
// MI455X (gfx1250) — matrix operand order repaired and checked by reading, not compiled, not run on hardware
//
#include <hip/hip_runtime.h>

typedef __attribute__((ext_vector_type(16))) _Float16 v16h;
typedef __attribute__((ext_vector_type(8)))  float    v8f;
typedef __attribute__((ext_vector_type(4)))  float    v4f;
typedef __attribute__((ext_vector_type(2)))  float    v2f;

#define W_    128
#define HW_   (128 * 128)
#define C_    64
#define OFFC_ 144   // DG(8) * 2 * K(9)

__device__ __forceinline__ int frame_of_pair(int p) {
    int j = p & 3;              // pairs 0..3 per batch -> frames 0,1,3,4 (skip center 2)
    return j < 2 ? j : j + 1;
}

__device__ __forceinline__ int clamp127(int v) {
    return min(max(v, 0), 127);
}

// ---------------------------------------------------------------------------
// LDS-tiled NCHW -> NHWC transpose of lrs (10 images of [64][HW] -> [HW][64]).
// ---------------------------------------------------------------------------
__global__ void transpose_to_nhwc(const float* __restrict__ in,
                                  float* __restrict__ out) {
    __shared__ float tile[32][33];
    const int bt = blockIdx.z;
    const float* src = in  + (size_t)bt * C_ * HW_;
    float*       dst = out + (size_t)bt * C_ * HW_;
    const int n0 = blockIdx.x * 32;
    const int c0 = blockIdx.y * 32;
#pragma unroll
    for (int i = threadIdx.y; i < 32; i += 8)
        tile[i][threadIdx.x] = src[(size_t)(c0 + i) * HW_ + n0 + threadIdx.x];
    __syncthreads();
#pragma unroll
    for (int i = threadIdx.y; i < 32; i += 8)
        dst[(size_t)(n0 + i) * C_ + c0 + threadIdx.x] = tile[threadIdx.x][i];
}

// ---------------------------------------------------------------------------
// Weight repack with tap-major k permutation:  k' = kk * Cin + ci.
// A 32-wide k-tile therefore has constant tap kk and 32 consecutive channels.
// Fragment f = (mt * KT + kt); lane holds 16 halfs in contiguous k order (same order used for B):
//   m = mt*16 + (lane&15);  ko = 16*(lane>=16) + e;
//   ci = (kt % (Cin/32))*32 + ko;  kk = kt / (Cin/32);
//   value = wgt[m][ci*9 + kk]   (OIHW, 3x3)
// ---------------------------------------------------------------------------
__global__ __launch_bounds__(32) void pack_weights(const float* __restrict__ wgt,
                                                   _Float16* __restrict__ dst,
                                                   int Cin) {
    const int lane = threadIdx.x;
    const int kt   = blockIdx.x;          // k-tile (32 wide)
    const int mt   = blockIdx.y;          // m-tile (16 wide)
    const int KT   = gridDim.x;
    const int Ktot = Cin * 9;
    const int tpc  = Cin / 32;            // k-tiles per tap
    const int kk   = kt / tpc;
    const int cib  = (kt - kk * tpc) * 32;
    const int m    = mt * 16 + (lane & 15);
    const int sel  = (lane >> 4) * 16;
    const float* wr = wgt + (size_t)m * Ktot;
    _Float16* out = dst + (((size_t)mt * KT + kt) * 32 + lane) * 16;
#pragma unroll
    for (int e = 0; e < 16; ++e) {
        int ko = sel + e;
        int ci = cib + ko;
        out[e] = (_Float16)wr[ci * 9 + kk];
    }
}

__device__ __forceinline__ v16h load_wfrag_packed(const _Float16* __restrict__ wp,
                                                  int KT, int mt, int kt, int lane) {
    return ((const v16h*)wp)[((size_t)(mt * KT + kt)) * 32 + lane];
}

// Pick element i of 16 contiguous floats held as four v4f.
__device__ __forceinline__ float pick16(const v4f& q0, const v4f& q1,
                                        const v4f& q2, const v4f& q3, int e) {
    return e < 4 ? q0[e] : e < 8 ? q1[e - 4] : e < 12 ? q2[e - 8] : q3[e - 12];
}

// ---------------------------------------------------------------------------
// Aggregation conv (NHWC): Cin = 128 = concat[center, feat] from transposed
// lrs, Cout = 64, 3x3, pad 1, + bias.  One wave per 16-pixel tile.
// Tap-major k: kt -> kk = kt>>2, channel base = (kt&3)*32.  The 16 channels
// of a lane never straddle the concat seam, so the frame select is a single
// branch-free integer cndmask.
// ---------------------------------------------------------------------------
__global__ __launch_bounds__(32) void conv_agg_wmma(const float* __restrict__ lrsT,
                                                    const _Float16* __restrict__ wp,
                                                    const float* __restrict__ bias,
                                                    float* __restrict__ dst) {
    const int lane = threadIdx.x;
    const int p    = blockIdx.z;
    const int b    = p >> 2;
    const int fr   = frame_of_pair(p);
    const int n0   = blockIdx.x * 16;
    const int ncol = n0 + (lane & 15);
    const int y0   = ncol >> 7;
    const int x0   = ncol & (W_ - 1);
    const int KT   = 36;                  // 128*9/32

    const float* lrsb = lrsT + (size_t)(b * 5) * C_ * HW_;

    v8f acc[4] = {};
    const int bsel = (lane >> 4) * 16;

#pragma unroll 1
    for (int kt = 0; kt < KT; ++kt) {
        const int kk = kt >> 2;
        const int kh = kk / 3, kw = kk - kh * 3;
        const int cb = ((kt & 3) << 5) + bsel;    // lane's first channel (16 consecutive)
        const int iy = y0 + kh - 1, ix = x0 + kw - 1;
        const bool valid = ((unsigned)iy < 128u) & ((unsigned)ix < 128u);
        const int idx = clamp127(iy) * W_ + clamp127(ix);
        const int tsel = (cb < C_) ? 2 : fr;      // frame index (branch-free select)
        const float* sp = lrsb + ((size_t)tsel * HW_ + idx) * C_ + (cb & 63);
        v4f q0 = *(const v4f*)sp;
        v4f q1 = *(const v4f*)(sp + 4);
        v4f q2 = *(const v4f*)(sp + 8);
        v4f q3 = *(const v4f*)(sp + 12);
        v16h bf;
#pragma unroll
        for (int e = 0; e < 16; ++e) {
            float x = pick16(q0, q1, q2, q3, e);
            bf[e] = (_Float16)(valid ? x : 0.f);
        }
#pragma unroll
        for (int mt = 0; mt < 4; ++mt) {
            v16h a = load_wfrag_packed(wp, KT, mt, kt, lane);
            acc[mt] = __builtin_amdgcn_wmma_f32_16x16x32_f16(
                false, a, false, bf, (short)0, acc[mt], false, false);
        }
    }

    // NHWC store: lane owns pixel ncol, 8 contiguous channels per m-tile half.
    const int rbase = (lane >> 4) * 8;
    float* drow = dst + ((size_t)p * HW_ + ncol) * C_;
#pragma unroll
    for (int mt = 0; mt < 4; ++mt) {
        v4f s0, s1;
#pragma unroll
        for (int r = 0; r < 4; ++r) {
            s0[r] = acc[mt][r]     + bias[mt * 16 + rbase + r];
            s1[r] = acc[mt][r + 4] + bias[mt * 16 + rbase + r + 4];
        }
        *(v4f*)(drow + mt * 16 + rbase)     = s0;
        *(v4f*)(drow + mt * 16 + rbase + 4) = s1;
    }
}

// ---------------------------------------------------------------------------
// Offset conv (NHWC): Cin = 64, Cout = 144, 3x3, pad 1, + bias.
// One wave per 16-pixel tile, 9 accumulators (144 = 9 * 16 exactly).
// ---------------------------------------------------------------------------
__global__ __launch_bounds__(32) void conv_off_wmma(const float* __restrict__ in,
                                                    const _Float16* __restrict__ wp,
                                                    const float* __restrict__ bias,
                                                    float* __restrict__ dst) {
    const int lane = threadIdx.x;
    const int p    = blockIdx.z;
    const int n0   = blockIdx.x * 16;
    const int ncol = n0 + (lane & 15);
    const int y0   = ncol >> 7;
    const int x0   = ncol & (W_ - 1);
    const int KT   = 18;                  // 64*9/32

    const float* src = in + (size_t)p * HW_ * C_;

    v8f acc[9] = {};
    const int bsel = (lane >> 4) * 16;

#pragma unroll 1
    for (int kt = 0; kt < KT; ++kt) {
        const int kk = kt >> 1;
        const int kh = kk / 3, kw = kk - kh * 3;
        const int cb = ((kt & 1) << 5) + bsel;
        const int iy = y0 + kh - 1, ix = x0 + kw - 1;
        const bool valid = ((unsigned)iy < 128u) & ((unsigned)ix < 128u);
        const int idx = clamp127(iy) * W_ + clamp127(ix);
        const float* sp = src + (size_t)idx * C_ + cb;
        v4f q0 = *(const v4f*)sp;
        v4f q1 = *(const v4f*)(sp + 4);
        v4f q2 = *(const v4f*)(sp + 8);
        v4f q3 = *(const v4f*)(sp + 12);
        v16h bf;
#pragma unroll
        for (int e = 0; e < 16; ++e) {
            float x = pick16(q0, q1, q2, q3, e);
            bf[e] = (_Float16)(valid ? x : 0.f);
        }
#pragma unroll
        for (int mt = 0; mt < 9; ++mt) {
            v16h a = load_wfrag_packed(wp, KT, mt, kt, lane);
            acc[mt] = __builtin_amdgcn_wmma_f32_16x16x32_f16(
                false, a, false, bf, (short)0, acc[mt], false, false);
        }
    }

    const int rbase = (lane >> 4) * 8;
    float* drow = dst + ((size_t)p * HW_ + ncol) * OFFC_;
#pragma unroll
    for (int mt = 0; mt < 9; ++mt) {
        v4f s0, s1;
#pragma unroll
        for (int r = 0; r < 4; ++r) {
            s0[r] = acc[mt][r]     + bias[mt * 16 + rbase + r];
            s1[r] = acc[mt][r + 4] + bias[mt * 16 + rbase + r + 4];
        }
        *(v4f*)(drow + mt * 16 + rbase)     = s0;
        *(v4f*)(drow + mt * 16 + rbase + 4) = s1;
    }
}

// ---------------------------------------------------------------------------
// Deformable conv (NHWC): K = 576, Cout = 64.  Tap-major k: each k-tile has
// one tap kk; a lane's 16 consecutive channels span exactly 2 deform groups,
// and all 8 channels of a group share one (dy,dx) and 4 bilinear corners.
// Per tile: 2 x b64 offset loads, 2 x 4 corners x 2 x b128 channel-vector
// loads, branch-free OOB (corner weight zeroed, index clamped).
// MODE 0: input from NHWC workspace buffer;  MODE 1: input = transposed feat.
// ---------------------------------------------------------------------------
template <int MODE>
__global__ __launch_bounds__(32) void deform_wmma(const float* __restrict__ in,
                                                  const float* __restrict__ lrsT,
                                                  const float* __restrict__ off,
                                                  const _Float16* __restrict__ wp,
                                                  float* __restrict__ dst) {
    const int lane = threadIdx.x;
    const int p    = blockIdx.z;
    const int b    = p >> 2;
    const int fr   = frame_of_pair(p);
    const int n0   = blockIdx.x * 16;
    const int ncol = n0 + (lane & 15);
    const int y0   = ncol >> 7;
    const int x0   = ncol & (W_ - 1);
    const int KT   = 18;

    const float* src = (MODE == 0) ? (in + (size_t)p * HW_ * C_)
                                   : (lrsT + (size_t)(b * 5 + fr) * HW_ * C_);
    const float* ob = off + ((size_t)p * HW_ + ncol) * OFFC_;

    v8f acc[4] = {};
    const int bsel = (lane >> 4) * 16;

#pragma unroll 1
    for (int kt = 0; kt < KT; ++kt) {
        const int kk = kt >> 1;
        const int kh = kk / 3, kw = kk - kh * 3;
        const int cb = ((kt & 1) << 5) + bsel;    // lane's first channel (mult of 16)

        v16h bf;
#pragma unroll
        for (int h = 0; h < 2; ++h) {
            const int g  = (cb >> 3) + h;         // deform group (Cg = 8)
            const int oc = (g * 9 + kk) * 2;      // dy at oc, dx at oc+1
            v2f d = *(const v2f*)(ob + oc);
            float py = (float)(y0 + kh - 1) + d[0];
            float px = (float)(x0 + kw - 1) + d[1];
            float fy = floorf(py), fx = floorf(px);
            float wy = py - fy,    wx = px - fx;
            int iy = (int)fy, ix = (int)fx;
            bool yv0 = (unsigned)iy       < 128u;
            bool yv1 = (unsigned)(iy + 1) < 128u;
            bool xv0 = (unsigned)ix       < 128u;
            bool xv1 = (unsigned)(ix + 1) < 128u;
            int iy0 = clamp127(iy), iy1 = clamp127(iy + 1);
            int ix0 = clamp127(ix), ix1 = clamp127(ix + 1);
            float w00 = (yv0 && xv0) ? (1.f - wy) * (1.f - wx) : 0.f;
            float w01 = (yv0 && xv1) ? (1.f - wy) * wx         : 0.f;
            float w10 = (yv1 && xv0) ? wy * (1.f - wx)         : 0.f;
            float w11 = (yv1 && xv1) ? wy * wx                 : 0.f;
            const float* gsrc = src + cb + h * 8;  // 8 contiguous channels
            const float* c00p = gsrc + (size_t)(iy0 * W_ + ix0) * C_;
            const float* c01p = gsrc + (size_t)(iy0 * W_ + ix1) * C_;
            const float* c10p = gsrc + (size_t)(iy1 * W_ + ix0) * C_;
            const float* c11p = gsrc + (size_t)(iy1 * W_ + ix1) * C_;
            v4f a0 = *(const v4f*)c00p,       a1 = *(const v4f*)(c00p + 4);
            v4f b0 = *(const v4f*)c01p,       b1 = *(const v4f*)(c01p + 4);
            v4f c0 = *(const v4f*)c10p,       c1 = *(const v4f*)(c10p + 4);
            v4f d0 = *(const v4f*)c11p,       d1 = *(const v4f*)(c11p + 4);
#pragma unroll
            for (int e2 = 0; e2 < 8; ++e2) {
                float va = e2 < 4 ? a0[e2] : a1[e2 - 4];
                float vb = e2 < 4 ? b0[e2] : b1[e2 - 4];
                float vc = e2 < 4 ? c0[e2] : c1[e2 - 4];
                float vd = e2 < 4 ? d0[e2] : d1[e2 - 4];
                float v = w00 * va + w01 * vb + w10 * vc + w11 * vd;
                bf[h * 8 + e2] = (_Float16)v;
            }
        }
#pragma unroll
        for (int mt = 0; mt < 4; ++mt) {
            v16h a = load_wfrag_packed(wp, KT, mt, kt, lane);
            acc[mt] = __builtin_amdgcn_wmma_f32_16x16x32_f16(
                false, a, false, bf, (short)0, acc[mt], false, false);
        }
    }

    const int rbase = (lane >> 4) * 8;
    float* drow = dst + ((size_t)p * HW_ + ncol) * C_;
#pragma unroll
    for (int mt = 0; mt < 4; ++mt) {
        v4f s0, s1;
#pragma unroll
        for (int r = 0; r < 4; ++r) { s0[r] = acc[mt][r]; s1[r] = acc[mt][r + 4]; }
        *(v4f*)(drow + mt * 16 + rbase)     = s0;
        *(v4f*)(drow + mt * 16 + rbase + 4) = s1;
    }
}

// ---------------------------------------------------------------------------
// Final assembly: out (NCHW) = lrs[b][2] for t==2, else NHWC result for pair.
// ---------------------------------------------------------------------------
__global__ void assemble_out(const float* __restrict__ lrs,
                             const float* __restrict__ res,
                             float* __restrict__ out, int total) {
    int i = blockIdx.x * blockDim.x + threadIdx.x;
    if (i >= total) return;
    int n = i % HW_;
    int c = (i / HW_) % C_;
    int t = (i / (HW_ * C_)) % 5;
    int b = i / (HW_ * C_ * 5);
    float v;
    if (t == 2) {
        v = lrs[i];
    } else {
        int j = t < 2 ? t : t - 1;
        int pr = b * 4 + j;
        v = res[((size_t)pr * HW_ + n) * C_ + c];
    }
    out[i] = v;
}

extern "C" void kernel_launch(void* const* d_in, const int* in_sizes, int n_in,
                              void* d_out, int out_size, void* d_ws, size_t ws_size,
                              hipStream_t stream) {
    (void)in_sizes; (void)n_in; (void)out_size; (void)ws_size;

    const float* lrs   = (const float*)d_in[0];
    const float* agg_w = (const float*)d_in[1];
    const float* agg_b = (const float*)d_in[2];
    const float* d1_ow = (const float*)d_in[3];
    const float* d1_ob = (const float*)d_in[4];
    const float* d1_w  = (const float*)d_in[5];
    const float* d2_ow = (const float*)d_in[6];
    const float* d2_ob = (const float*)d_in[7];
    const float* d2_w  = (const float*)d_in[8];
    const float* a1_ow = (const float*)d_in[9];
    const float* a1_ob = (const float*)d_in[10];
    const float* a1_w  = (const float*)d_in[11];
    const float* a2_ow = (const float*)d_in[12];
    const float* a2_ob = (const float*)d_in[13];
    const float* a2_w  = (const float*)d_in[14];

    float* ws   = (float*)d_ws;
    float* lrsT = ws;                                  // 10 * 64  * HW floats (NHWC)
    float* bufA = lrsT + (size_t)10 * C_ * HW_;        //  8 * HW * 64 floats (NHWC)
    float* bufB = bufA + (size_t)8 * C_ * HW_;         //  8 * HW * 64 floats (NHWC)
    float* offb = bufB + (size_t)8 * C_ * HW_;         //  8 * HW * 144 floats (NHWC)

    // Packed f16 weight fragments (after offb; ~1.1 MB total, 32B aligned).
    _Float16* wp      = (_Float16*)(offb + (size_t)8 * OFFC_ * HW_);
    const size_t AGG_SZ = (size_t)4 * 36 * 32 * 16;    // 73728 halfs
    const size_t OFF_SZ = (size_t)9 * 18 * 32 * 16;    // 82944 halfs
    const size_t DEF_SZ = (size_t)4 * 18 * 32 * 16;    // 36864 halfs
    _Float16* wp_agg = wp;
    _Float16* wp_o1  = wp_agg + AGG_SZ;
    _Float16* wp_o2  = wp_o1  + OFF_SZ;
    _Float16* wp_o3  = wp_o2  + OFF_SZ;
    _Float16* wp_o4  = wp_o3  + OFF_SZ;
    _Float16* wp_d1  = wp_o4  + OFF_SZ;
    _Float16* wp_d2  = wp_d1  + DEF_SZ;
    _Float16* wp_d3  = wp_d2  + DEF_SZ;
    _Float16* wp_d4  = wp_d3  + DEF_SZ;

    // One-time layout conversion + weight repack (stream-ordered).
    transpose_to_nhwc<<<dim3(HW_ / 32, C_ / 32, 10), dim3(32, 8), 0, stream>>>(lrs, lrsT);
    pack_weights<<<dim3(36, 4), 32, 0, stream>>>(agg_w, wp_agg, 128);
    pack_weights<<<dim3(18, 9), 32, 0, stream>>>(d1_ow, wp_o1, C_);
    pack_weights<<<dim3(18, 9), 32, 0, stream>>>(d2_ow, wp_o2, C_);
    pack_weights<<<dim3(18, 9), 32, 0, stream>>>(a1_ow, wp_o3, C_);
    pack_weights<<<dim3(18, 9), 32, 0, stream>>>(a2_ow, wp_o4, C_);
    pack_weights<<<dim3(18, 4), 32, 0, stream>>>(d1_w,  wp_d1, C_);
    pack_weights<<<dim3(18, 4), 32, 0, stream>>>(d2_w,  wp_d2, C_);
    pack_weights<<<dim3(18, 4), 32, 0, stream>>>(a1_w,  wp_d3, C_);
    pack_weights<<<dim3(18, 4), 32, 0, stream>>>(a2_w,  wp_d4, C_);

    dim3 grid(HW_ / 16, 1, 8);                          // 1024 pixel tiles x 8 (b,frame) pairs

    conv_agg_wmma<<<grid, 32, 0, stream>>>(lrsT, wp_agg, agg_b, bufA);

    conv_off_wmma<<<grid, 32, 0, stream>>>(bufA, wp_o1, d1_ob, offb);
    deform_wmma<0><<<grid, 32, 0, stream>>>(bufA, lrsT, offb, wp_d1, bufB);

    conv_off_wmma<<<grid, 32, 0, stream>>>(bufB, wp_o2, d2_ob, offb);
    deform_wmma<0><<<grid, 32, 0, stream>>>(bufB, lrsT, offb, wp_d2, bufA);

    conv_off_wmma<<<grid, 32, 0, stream>>>(bufA, wp_o3, a1_ob, offb);
    deform_wmma<1><<<grid, 32, 0, stream>>>(nullptr, lrsT, offb, wp_d3, bufB);

    conv_off_wmma<<<grid, 32, 0, stream>>>(bufB, wp_o4, a2_ob, offb);
    deform_wmma<0><<<grid, 32, 0, stream>>>(bufB, lrsT, offb, wp_d4, bufA);

    int total = 2 * 5 * C_ * HW_;
    assemble_out<<<(total + 255) / 256, 256, 0, stream>>>(lrs, bufA, (float*)d_out, total);
}
